// AttentionDistillationLoss_4698694222571
// MI455X (gfx1250) — matrix operand order repaired and checked by reading, not compiled, not run on hardware
//
#include <hip/hip_runtime.h>
#include <hip/hip_bf16.h>

typedef _Float16 half_t;
typedef __attribute__((ext_vector_type(16))) _Float16 v16h;
typedef __attribute__((ext_vector_type(8)))  float    v8f;

#define C_    128
#define M_    256
#define K_    32
#define NEG_  (-1e9f)

// ---------------------------------------------------------------------------
// Kernel 1: for each gather slot (M node rows + M*K neighbor rows), compute
// sf = exp(x - max) / ||exp(x - max)||  (softmax + L2-normalize, denominators
// cancel) and store as f16. One wave32 per row, 4 elements per lane.
// ---------------------------------------------------------------------------
__global__ void adl_softmax_norm_kernel(const float* __restrict__ student,
                                        const int* __restrict__ node_ids,
                                        const int* __restrict__ nbr_idx,
                                        half_t* __restrict__ feat,
                                        int total_rows) {
    int wave = (int)((blockIdx.x * blockDim.x + threadIdx.x) >> 5);
    int lane = (int)(threadIdx.x & 31);
    if (wave >= total_rows) return;

    int idx = (wave < M_) ? node_ids[wave] : nbr_idx[wave - M_];
    const float* row = student + (size_t)idx * C_;

    float x[4];
#pragma unroll
    for (int i = 0; i < 4; ++i) x[i] = row[lane + 32 * i];

    float mx = fmaxf(fmaxf(x[0], x[1]), fmaxf(x[2], x[3]));
#pragma unroll
    for (int o = 16; o > 0; o >>= 1) mx = fmaxf(mx, __shfl_xor(mx, o, 32));

    float e[4];
    float ss = 0.0f;
#pragma unroll
    for (int i = 0; i < 4; ++i) { e[i] = __expf(x[i] - mx); ss += e[i] * e[i]; }
#pragma unroll
    for (int o = 16; o > 0; o >>= 1) ss += __shfl_xor(ss, o, 32);

    float inv = rsqrtf(ss);
    half_t* dst = feat + (size_t)wave * C_;
#pragma unroll
    for (int i = 0; i < 4; ++i) dst[lane + 32 * i] = (half_t)(e[i] * inv);
}

// ---------------------------------------------------------------------------
// Kernel 2: one wave32 per sampled node m. sim[32] = nbr[32x128] . node[128]
// via 8x v_wmma_f32_16x16x32_f16 (2 neighbor groups x 4 K-chunks; B columns
// replicated with the node vector, so D[i,0] = dot). Then masked log-softmax
// over the 32 neighbors + KL(teacher || student), one neighbor per lane.
// ---------------------------------------------------------------------------
__global__ void adl_sim_kl_kernel(const half_t* __restrict__ feat,
                                  const float* __restrict__ teacher_w,
                                  const unsigned char* __restrict__ nmask,
                                  float* __restrict__ kl_out) {
    __shared__ float sims[K_];
    const int m    = (int)blockIdx.x;
    const int lane = (int)threadIdx.x;          // 0..31, one full wave
    const bool hi  = lane >= 16;
    const int lrow = lane & 15;

    const half_t* node = feat + (size_t)m * C_;
    const half_t* nbr  = feat + (size_t)(M_ + m * K_) * C_;

#pragma unroll
    for (int g = 0; g < 2; ++g) {
        v8f acc = {};
#pragma unroll
        for (int t = 0; t < 4; ++t) {
            // A (16x32 f16): lanes 0-15 -> K 0..15 of row 16g+lrow,
            //                lanes 16-31 -> K 16..31 of row 16g+lrow.
            const half_t* arow = nbr + (size_t)(16 * g + lrow) * C_ + 32 * t + (hi ? 16 : 0);
            v16h A;
#pragma unroll
            for (int i = 0; i < 8; ++i) { A[i] = arow[i]; A[i + 8] = arow[i + 8]; }

            // B (32x16 f16), all 16 columns = node chunk:
            // lanes 0-15 hold K 0..15, lanes 16-31 hold K 16..31.
            const half_t* brow = node + 32 * t + (hi ? 16 : 0);
            v16h B;
#pragma unroll
            for (int i = 0; i < 16; ++i) B[i] = brow[i];

            acc = __builtin_amdgcn_wmma_f32_16x16x32_f16(
                false, A, false, B, (short)0, acc, false, false);
        }
        // D layout: lane j<16: VGPR r = D[M=r, N=j]; lane j>=16: VGPR r = D[M=8+r].
        if (lane == 0) {
#pragma unroll
            for (int r = 0; r < 8; ++r) sims[16 * g + r] = acc[r];
        }
        if (lane == 16) {
#pragma unroll
            for (int r = 0; r < 8; ++r) sims[16 * g + 8 + r] = acc[r];
        }
    }
    __syncthreads();

    const float sim   = sims[lane];
    const bool  valid = nmask[(size_t)m * K_ + lane] != 0;
    const float s_lg  = valid ? sim : NEG_;                              // TEMP = 1
    const float t_lg  = valid ? teacher_w[(size_t)m * K_ + lane] : NEG_;

    float smax = s_lg, tmax = t_lg;
#pragma unroll
    for (int o = 16; o > 0; o >>= 1) {
        smax = fmaxf(smax, __shfl_xor(smax, o, 32));
        tmax = fmaxf(tmax, __shfl_xor(tmax, o, 32));
    }
    float se = __expf(s_lg - smax), te = __expf(t_lg - tmax);
    float ssum = se, tsum = te;
#pragma unroll
    for (int o = 16; o > 0; o >>= 1) {
        ssum += __shfl_xor(ssum, o, 32);
        tsum += __shfl_xor(tsum, o, 32);
    }
    const float log_s = s_lg - smax - __logf(ssum);
    const float log_t = t_lg - tmax - __logf(tsum);
    const float tdist = __expf(log_t);

    float term = valid ? tdist * (log_t - log_s) : 0.0f;
#pragma unroll
    for (int o = 16; o > 0; o >>= 1) term += __shfl_xor(term, o, 32);

    if (lane == 0) kl_out[m] = term;
}

// ---------------------------------------------------------------------------
// Kernel 3: deterministic tree reduction of the M per-node KL values,
// write mean into d_out[0] (fully overwrites the poisoned output).
// ---------------------------------------------------------------------------
__global__ void adl_reduce_kernel(const float* __restrict__ kl, float* __restrict__ out) {
    __shared__ float buf[M_];
    const int t = (int)threadIdx.x;
    buf[t] = kl[t];
    __syncthreads();
#pragma unroll
    for (int s = M_ / 2; s > 0; s >>= 1) {
        if (t < s) buf[t] += buf[t + s];
        __syncthreads();
    }
    if (t == 0) out[0] = buf[0] / (float)M_;
}

extern "C" void kernel_launch(void* const* d_in, const int* in_sizes, int n_in,
                              void* d_out, int out_size, void* d_ws, size_t ws_size,
                              hipStream_t stream) {
    const float*         student   = (const float*)d_in[0];         // [N, C] f32
    const float*         teacher_w = (const float*)d_in[1];         // [M, K] f32
    const int*           node_ids  = (const int*)d_in[2];           // [M] i32
    const int*           nbr_idx   = (const int*)d_in[3];           // [M, K] i32
    const unsigned char* nmask     = (const unsigned char*)d_in[4]; // [M, K] bool (1B)
    float*               out       = (float*)d_out;

    (void)in_sizes; (void)n_in; (void)out_size; (void)ws_size;

    // Workspace layout: f16 features for all gather slots, then per-node KL.
    const int total_rows = M_ + M_ * K_;                        // 8448
    half_t* feat = (half_t*)d_ws;
    float*  kl   = (float*)((char*)d_ws + (size_t)total_rows * C_ * sizeof(half_t));

    // K1: gather + softmax + L2-normalize -> f16 features (1 wave per row).
    {
        const int threads = 256;                                // 8 waves/block
        const int blocks  = (total_rows * 32 + threads - 1) / threads;
        adl_softmax_norm_kernel<<<blocks, threads, 0, stream>>>(
            student, node_ids, nbr_idx, feat, total_rows);
    }
    // K2: WMMA cosine sims + masked KL (1 wave per sampled node).
    adl_sim_kl_kernel<<<M_, 32, 0, stream>>>(feat, teacher_w, nmask, kl);

    // K3: mean over M.
    adl_reduce_kernel<<<1, M_, 0, stream>>>(kl, out);
}
